// ResidualBlock_basic_71193377898713
// MI455X (gfx1250) — matrix operand order repaired and checked by reading, not compiled, not run on hardware
//
#include <hip/hip_runtime.h>
#include <hip/hip_bf16.h>
#include <stdint.h>

#define NVOX 100000
#define CCH  64
#define KOFF 27
#define EPSV 1e-4f
#define WSWZ (KOFF * 2 * 4 * 32 * 16)   // swizzled weight elements per conv
#define NTILE (NVOX / 16)               // 6250 M-tiles, exact
#define NGRP  (NTILE / 2)               // 3125 wave-tasks (2 M-tiles each), exact

typedef __attribute__((ext_vector_type(16))) __bf16 v16bf;
typedef __attribute__((ext_vector_type(8)))  float  v8f;

union Frag { v16bf v; uint4 q[2]; };

__device__ __forceinline__ unsigned short f2bf(float f) {
    union { float f; unsigned u; } x; x.f = f;
    unsigned u = x.u;
    u += 0x7FFFu + ((u >> 16) & 1u);       // round-to-nearest-even
    return (unsigned short)(u >> 16);
}

// ---- BN + ReLU on f32 features -> bf16 activations --------------------------
__global__ void bn_relu_bf16_kernel(const float* __restrict__ x,
                                    const float* __restrict__ g,
                                    const float* __restrict__ b,
                                    const float* __restrict__ m,
                                    const float* __restrict__ v,
                                    unsigned short* __restrict__ y) {
    int i = blockIdx.x * blockDim.x + threadIdx.x;
    if (i >= NVOX * CCH) return;
    int ch = i & (CCH - 1);
    float scale = g[ch] * rsqrtf(v[ch] + EPSV);
    float shift = b[ch] - m[ch] * scale;
    float val = fmaf(x[i], scale, shift);
    y[i] = f2bf(val > 0.f ? val : 0.f);
}

// ---- pre-swizzle f32 weights [27][C][C] into bf16 B-fragment layout ---------
// dest[k][s][n][lane][j]:
//   lane<16 : column d = 16n+lane,     K = 32s + j        (j = 0..15)
//   lane>=16: column d = 16n+lane-16,  K = 32s + 16 + j
__global__ void swizzle_w_kernel(const float* __restrict__ W,
                                 unsigned short* __restrict__ Wsw) {
    int t = blockIdx.x * blockDim.x + threadIdx.x;
    if (t >= WSWZ) return;
    int j    =  t        & 15;
    int lane = (t >> 4)  & 31;
    int n    = (t >> 9)  & 3;
    int s    = (t >> 11) & 1;
    int k    =  t >> 12;
    int d = 16 * n + (lane & 15);
    int c = 32 * s + (lane < 16 ? j : 16 + j);
    Wsw[t] = f2bf(W[(k * CCH + c) * CCH + d]);
}

// ---- WMMA submanifold conv: one wave = 2 M-tiles (32 voxels) x 64 channels --
// Each B fragment is loaded once and reused for both M-tiles (halves weight
// traffic per WMMA). EPI==0: fuse BN2+ReLU, store bf16. EPI==1: residual, f32.
template <int EPI>
__global__ __launch_bounds__(256, 2)
void subm_conv_wmma_kernel(const unsigned short* __restrict__ X,   // bf16 [N][64]
                           const int* __restrict__ nbr,            // [N][27]
                           const unsigned short* __restrict__ Wsw, // swizzled bf16
                           const float* __restrict__ g,  const float* __restrict__ b,
                           const float* __restrict__ m,  const float* __restrict__ v,
                           const float* __restrict__ resid,
                           void* __restrict__ out) {
    const int lane = threadIdx.x & 31;
    const int wave = threadIdx.x >> 5;
    const int grp  = blockIdx.x * 8 + wave;
    if (grp >= NGRP) return;                       // wave-uniform; EXEC stays full
    const int hi   = lane >> 4;
    const int base = grp * 32 + (lane & 15);       // voxel of M-tile 0 for this lane

    v8f acc[2][4];
#pragma unroll
    for (int t = 0; t < 2; ++t)
#pragma unroll
        for (int n = 0; n < 4; ++n)
            acc[t][n] = (v8f){0.f, 0.f, 0.f, 0.f, 0.f, 0.f, 0.f, 0.f};

#pragma unroll 1
    for (int k = 0; k < KOFF; ++k) {
        if (k + 1 < KOFF)   // keep next offset's weight block streaming into cache
            __builtin_prefetch(Wsw + ((size_t)((k + 1) * 8) * 32) * 16, 0, 1);

        int idx[2];
        bool valid[2];
#pragma unroll
        for (int t = 0; t < 2; ++t) {
            idx[t]   = nbr[(base + 16 * t) * KOFF + k];
            valid[t] = idx[t] >= 0;
        }
#pragma unroll
        for (int s = 0; s < 2; ++s) {
            // A fragments: lanes 0-15 -> K 32s+0..15; lanes 16-31 -> K 32s+16..31
            Frag a[2];
#pragma unroll
            for (int t = 0; t < 2; ++t) {
                const unsigned short* p =
                    X + (size_t)(valid[t] ? idx[t] : 0) * CCH + 32 * s + 16 * hi;
                a[t].q[0] = *(const uint4*)(p);
                a[t].q[1] = *(const uint4*)(p + 8);
                if (!valid[t]) {
                    a[t].q[0] = make_uint4(0, 0, 0, 0);
                    a[t].q[1] = make_uint4(0, 0, 0, 0);
                }
            }
            const unsigned short* wb =
                Wsw + ((size_t)((k * 2 + s) * 4) * 32 + lane) * 16;
#pragma unroll
            for (int n = 0; n < 4; ++n) {
                Frag bf;
                const unsigned short* wp = wb + (size_t)n * (32 * 16);
                bf.q[0] = *(const uint4*)(wp);
                bf.q[1] = *(const uint4*)(wp + 8);
#pragma unroll
                for (int t = 0; t < 2; ++t)   // reuse B for both M-tiles
                    acc[t][n] = __builtin_amdgcn_wmma_f32_16x16x32_bf16(
                        false, a[t].v, false, bf.v, (short)0, acc[t][n], false, false);
            }
        }
    }

    // Epilogue: C/D layout — VGPR j : M = j (lanes 0-15) / M = 8+j (lanes 16-31)
#pragma unroll
    for (int n = 0; n < 4; ++n) {
        const int ch = 16 * n + (lane & 15);
        if (EPI == 0) {
            float scale = g[ch] * rsqrtf(v[ch] + EPSV);
            float shift = b[ch] - m[ch] * scale;
            unsigned short* y = (unsigned short*)out;
#pragma unroll
            for (int t = 0; t < 2; ++t)
#pragma unroll
                for (int j = 0; j < 8; ++j) {
                    int mrow = grp * 32 + 16 * t + hi * 8 + j;
                    float val = fmaf(acc[t][n][j], scale, shift);
                    y[(size_t)mrow * CCH + ch] = f2bf(val > 0.f ? val : 0.f);
                }
        } else {
            float* y = (float*)out;
#pragma unroll
            for (int t = 0; t < 2; ++t)
#pragma unroll
                for (int j = 0; j < 8; ++j) {
                    int mrow = grp * 32 + 16 * t + hi * 8 + j;
                    y[(size_t)mrow * CCH + ch] =
                        resid[(size_t)mrow * CCH + ch] + acc[t][n][j];
                }
        }
    }
}

extern "C" void kernel_launch(void* const* d_in, const int* in_sizes, int n_in,
                              void* d_out, int out_size, void* d_ws, size_t ws_size,
                              hipStream_t stream) {
    const float* features = (const float*)d_in[0];
    const int*   nbr      = (const int*)d_in[1];
    const float* g1 = (const float*)d_in[2];
    const float* b1 = (const float*)d_in[3];
    const float* m1 = (const float*)d_in[4];
    const float* v1 = (const float*)d_in[5];
    const float* W1 = (const float*)d_in[6];
    const float* g2 = (const float*)d_in[7];
    const float* b2 = (const float*)d_in[8];
    const float* m2 = (const float*)d_in[9];
    const float* v2 = (const float*)d_in[10];
    const float* W2 = (const float*)d_in[11];

    // workspace layout (bf16): y1[N*64] | y2[N*64] | W1sw | W2sw  (~26 MB)
    unsigned short* y1  = (unsigned short*)d_ws;
    unsigned short* y2  = y1 + (size_t)NVOX * CCH;
    unsigned short* w1s = y2 + (size_t)NVOX * CCH;
    unsigned short* w2s = w1s + WSWZ;

    swizzle_w_kernel<<<(WSWZ + 255) / 256, 256, 0, stream>>>(W1, w1s);
    swizzle_w_kernel<<<(WSWZ + 255) / 256, 256, 0, stream>>>(W2, w2s);

    bn_relu_bf16_kernel<<<(NVOX * CCH + 255) / 256, 256, 0, stream>>>(
        features, g1, b1, m1, v1, y1);

    const int nblk = (NGRP + 7) / 8;               // 8 wave-tasks per 256-thread block

    subm_conv_wmma_kernel<0><<<nblk, 256, 0, stream>>>(
        y1, nbr, w1s, g2, b2, m2, v2, nullptr, (void*)y2);

    subm_conv_wmma_kernel<1><<<nblk, 256, 0, stream>>>(
        y2, nbr, w2s, nullptr, nullptr, nullptr, nullptr, features, d_out);
}
